// MyLoss_30631706755178
// MI455X (gfx1250) — matrix operand order repaired and checked by reading, not compiled, not run on hardware
//
#include <hip/hip_runtime.h>
#include <hip/hip_bf16.h>

typedef __bf16 bf16_t;
typedef __attribute__((ext_vector_type(16))) __bf16 v16bf;
typedef __attribute__((ext_vector_type(8)))  float  v8f;

#define DDIM 128

// ---------------------------------------------------------------------------
// Kernel 1: per-row squared norms + fp32 -> bf16 (hi, lo) split.
// One wave32 per row of length 128 (4 floats per lane).
// ---------------------------------------------------------------------------
__global__ __launch_bounds__(256)
void prep_kernel(const float* __restrict__ sub_x,
                 const float* __restrict__ all_x,
                 float* __restrict__ nx, float* __restrict__ ny,
                 bf16_t* __restrict__ xhi, bf16_t* __restrict__ xlo,
                 bf16_t* __restrict__ yhi, bf16_t* __restrict__ ylo,
                 int m, int n) {
    const int wave = (int)((blockIdx.x * blockDim.x + threadIdx.x) >> 5);
    const int lane = (int)(threadIdx.x & 31);
    if (wave >= m + n) return;

    const float* src;
    bf16_t *dh, *dl;
    float *dst_norm;
    if (wave < m) {
        src = sub_x + (size_t)wave * DDIM;
        dh = xhi + (size_t)wave * DDIM;
        dl = xlo + (size_t)wave * DDIM;
        dst_norm = nx + wave;
    } else {
        const int r = wave - m;
        src = all_x + (size_t)r * DDIM;
        dh = yhi + (size_t)r * DDIM;
        dl = ylo + (size_t)r * DDIM;
        dst_norm = ny + r;
    }

    const int base = lane * 4;
    const float4 v = *(const float4*)(src + base);
    float nrm = v.x * v.x + v.y * v.y + v.z * v.z + v.w * v.w;

    float elems[4] = {v.x, v.y, v.z, v.w};
#pragma unroll
    for (int k = 0; k < 4; ++k) {
        const bf16_t h = (bf16_t)elems[k];
        const bf16_t l = (bf16_t)(elems[k] - (float)h);
        dh[base + k] = h;
        dl[base + k] = l;
    }

#pragma unroll
    for (int off = 16; off > 0; off >>= 1)
        nrm += __shfl_xor(nrm, off, 32);
    if (lane == 0) *dst_norm = nrm;
}

// ---------------------------------------------------------------------------
// Kernel 2: fused  Σ G[i,j] * (nx_i + ny_j - 2 * x_i·y_j)  over 16x16 tiles.
// One wave per (i-tile, strip of JSTRIP j-tiles).
// S tile = X·Y^T via bf16x3 WMMA (hi*hi + hi*lo + lo*hi), K = 128 in 4 chunks.
// ---------------------------------------------------------------------------
template <int JSTRIP>
__global__ __launch_bounds__(256)
void cross_kernel(const float* __restrict__ G,
                  const bf16_t* __restrict__ xhi, const bf16_t* __restrict__ xlo,
                  const bf16_t* __restrict__ yhi, const bf16_t* __restrict__ ylo,
                  const float* __restrict__ nx, const float* __restrict__ ny,
                  float* __restrict__ partials, int m, int n) {
    const int lane  = (int)(threadIdx.x & 31);
    const int warp  = (int)(threadIdx.x >> 5);
    const int wave  = (int)(blockIdx.x * (blockDim.x >> 5) + warp);
    const int TJ     = n >> 4;
    const int strips = TJ / JSTRIP;
    const int itile  = wave / strips;
    const int strip  = wave % strips;
    const int i0     = itile << 4;
    const int jt0    = strip * JSTRIP;
    const int lrow   = lane & 15;   // row (A) / col (B,C) index within tile
    const int lhalf  = lane >> 4;   // which lane-half

    // ---- A fragments: 16x128 X tile (hi and lo), ISA 16-bit A layout ----
    // lane holds row (lane%16); K blocks of 8 selected by lane/16.
    v16bf ah[4], al[4];
    {
        const bf16_t* xh = xhi + (size_t)(i0 + lrow) * DDIM;
        const bf16_t* xl = xlo + (size_t)(i0 + lrow) * DDIM;
#pragma unroll
        for (int c = 0; c < 4; ++c) {
            const int b0 = c * 32 + lhalf * 8;
#pragma unroll
            for (int e = 0; e < 8; ++e) {
                ah[c][e]     = xh[b0 + e];
                ah[c][e + 8] = xh[b0 + 16 + e];
                al[c][e]     = xl[b0 + e];
                al[c][e + 8] = xl[b0 + 16 + e];
            }
        }
    }

    // nx values matching C layout: VGPR r -> M = r + 8*lhalf
    float nxr[8];
#pragma unroll
    for (int r = 0; r < 8; ++r) nxr[r] = nx[i0 + lhalf * 8 + r];

    float acc = 0.0f;
    for (int jt = jt0; jt < jt0 + JSTRIP; ++jt) {
        const int j0 = jt << 4;
        // B fragments: lane holds column (lane%16); K block by lane/16.
        const bf16_t* yh = yhi + (size_t)(j0 + lrow) * DDIM + lhalf * 8;
        const bf16_t* yl = ylo + (size_t)(j0 + lrow) * DDIM + lhalf * 8;

        v8f C = {0.0f, 0.0f, 0.0f, 0.0f, 0.0f, 0.0f, 0.0f, 0.0f};
#pragma unroll
        for (int c = 0; c < 4; ++c) {
            typedef __attribute__((ext_vector_type(8))) __bf16 v8bf_t;
            const v8bf_t bh0 = *(const v8bf_t*)(yh + c * 32);
            const v8bf_t bh1 = *(const v8bf_t*)(yh + c * 32 + 16);
            const v8bf_t bl0 = *(const v8bf_t*)(yl + c * 32);
            const v8bf_t bl1 = *(const v8bf_t*)(yl + c * 32 + 16);
            const v16bf bh = __builtin_shufflevector(bh0, bh1, 0, 1, 2, 3, 4, 5, 6, 7, 8, 9, 10, 11, 12, 13, 14, 15);
            const v16bf bl = __builtin_shufflevector(bl0, bl1, 0, 1, 2, 3, 4, 5, 6, 7, 8, 9, 10, 11, 12, 13, 14, 15);
            C = __builtin_amdgcn_wmma_f32_16x16x32_bf16(false, ah[c], false, bh,
                                                        (short)0, C, false, false);
            C = __builtin_amdgcn_wmma_f32_16x16x32_bf16(false, ah[c], false, bl,
                                                        (short)0, C, false, false);
            C = __builtin_amdgcn_wmma_f32_16x16x32_bf16(false, al[c], false, bh,
                                                        (short)0, C, false, false);
        }

        // Fused weighted reduction against the streamed G tile (non-temporal).
        const float nyv = ny[j0 + lrow];
        const float* g = G + (size_t)(i0 + lhalf * 8) * n + j0 + lrow;
        float tacc = 0.0f;
#pragma unroll
        for (int r = 0; r < 8; ++r) {
            const float gv = __builtin_nontemporal_load(g + (size_t)r * n);
            tacc += gv * (nxr[r] + nyv - 2.0f * C[r]);
        }
        acc += tacc;
    }

    // wave32 reduce -> block reduce -> deterministic per-block partial
#pragma unroll
    for (int off = 16; off > 0; off >>= 1)
        acc += __shfl_xor(acc, off, 32);

    __shared__ float sred[8];
    if (lane == 0) sred[warp] = acc;
    __syncthreads();
    if (threadIdx.x == 0) {
        float t = 0.0f;
        const int nw = (int)(blockDim.x >> 5);
        for (int w = 0; w < nw; ++w) t += sred[w];
        partials[blockIdx.x] = t;
    }
}

// ---------------------------------------------------------------------------
// Kernel 3: reduce block partials, scale by 1/(m*n).
// ---------------------------------------------------------------------------
__global__ __launch_bounds__(256)
void finish_kernel(const float* __restrict__ partials, int np,
                   float* __restrict__ out, float inv_mn) {
    __shared__ float s[256];
    float t = 0.0f;
    for (int i = (int)threadIdx.x; i < np; i += 256) t += partials[i];
    s[threadIdx.x] = t;
    __syncthreads();
#pragma unroll
    for (int off = 128; off > 0; off >>= 1) {
        if ((int)threadIdx.x < off) s[threadIdx.x] += s[threadIdx.x + off];
        __syncthreads();
    }
    if (threadIdx.x == 0) out[0] = s[0] * inv_mn;
}

// ---------------------------------------------------------------------------
extern "C" void kernel_launch(void* const* d_in, const int* in_sizes, int n_in,
                              void* d_out, int out_size, void* d_ws, size_t ws_size,
                              hipStream_t stream) {
    (void)n_in; (void)out_size; (void)ws_size;
    const float* G     = (const float*)d_in[0];   // [m, n]
    const float* sub_x = (const float*)d_in[1];   // [m, 128]
    const float* all_x = (const float*)d_in[2];   // [n, 128]
    const int m = in_sizes[1] / DDIM;             // 4096
    const int n = in_sizes[2] / DDIM;             // 8192

    // Workspace layout (all offsets 16B aligned for these sizes)
    char* ws = (char*)d_ws;
    float* nx = (float*)ws;                 ws += sizeof(float) * (size_t)m;
    float* ny = (float*)ws;                 ws += sizeof(float) * (size_t)n;
    bf16_t* xhi = (bf16_t*)ws;              ws += sizeof(bf16_t) * (size_t)m * DDIM;
    bf16_t* xlo = (bf16_t*)ws;              ws += sizeof(bf16_t) * (size_t)m * DDIM;
    bf16_t* yhi = (bf16_t*)ws;              ws += sizeof(bf16_t) * (size_t)n * DDIM;
    bf16_t* ylo = (bf16_t*)ws;              ws += sizeof(bf16_t) * (size_t)n * DDIM;
    float* partials = (float*)ws;

    // 1) norms + bf16 hi/lo split (one wave per row, 8 waves per block)
    const int rows = m + n;
    const int prep_blocks = (rows + 7) / 8;
    prep_kernel<<<prep_blocks, 256, 0, stream>>>(sub_x, all_x, nx, ny,
                                                 xhi, xlo, yhi, ylo, m, n);

    // 2) main fused pass: 1 wave per (i-tile, 16 j-tiles)
    constexpr int JSTRIP = 16;
    const int TI = m >> 4;                  // 256
    const int TJ = n >> 4;                  // 512
    const int strips = TJ / JSTRIP;         // 32
    const int total_waves = TI * strips;    // 8192
    const int blocks = total_waves / 8;     // 1024 blocks of 8 waves
    cross_kernel<JSTRIP><<<blocks, 256, 0, stream>>>(G, xhi, xlo, yhi, ylo,
                                                     nx, ny, partials, m, n);

    // 3) final reduction + scaling
    const float inv_mn = 1.0f / ((float)m * (float)n);
    finish_kernel<<<1, 256, 0, stream>>>(partials, blocks, (float*)d_out, inv_mn);
}
